// CausalSelfAttention_32049045962821
// MI455X (gfx1250) — compile-verified, matrix operand order checked by reading, not run on hardware
//
#include <hip/hip_runtime.h>
#include <hip/hip_bf16.h>
#include <stdint.h>

// Problem constants (reference: B=2, N=2048, D=1024, H=16, DH=64)
#define BB 2
#define NN 2048
#define DD 1024
#define HH 16
#define DH 64
#define SCALE 0.125f   // 1/sqrt(64)

typedef __bf16 bf16_t;
typedef bf16_t bf16x8  __attribute__((ext_vector_type(8)));
typedef bf16_t bf16x16 __attribute__((ext_vector_type(16)));
typedef float  f32x8   __attribute__((ext_vector_type(8)));

static __device__ __forceinline__ unsigned short f2bf(float f) {
  union { float f; unsigned u; } v; v.f = f;
  unsigned u = v.u;
  unsigned r = (u + 0x7FFFu + ((u >> 16) & 1u)) >> 16;  // RNE
  return (unsigned short)r;
}
static __device__ __forceinline__ bf16_t us2bf(unsigned short s) {
  union { unsigned short u; bf16_t b; } v; v.u = s; return v.b;
}

static __device__ __forceinline__ f32x8 wmma_bf16(bf16x16 a, bf16x16 b, f32x8 c) {
  // 8 args: (neg_a, A, neg_b, B, c_mod, C, reuse_a, reuse_b)
  return __builtin_amdgcn_wmma_f32_16x16x32_bf16(false, a, false, b, (short)0, c,
                                                 false, false);
}

// 16-bit A/B fragment (16x32), rows contiguous along k.
// ISA layout: lane L (r=L&15, half=L>>4): elems 0..7 = k kb..kb+7, elems 8..15 = kb+16..kb+23, kb=half*8.
// base points at [row0, k0]; stride in ushorts (must be multiple of 8 for 16B alignment).
static __device__ __forceinline__ bf16x16 load_frag_rowk(const unsigned short* base,
                                                         int stride, int lane) {
  int r  = lane & 15;
  int kb = (lane >> 4) * 8;
  const unsigned short* p = base + (size_t)r * stride + kb;
  bf16x8 lo = *(const bf16x8*)p;
  bf16x8 hi = *(const bf16x8*)(p + 16);
  bf16x16 a;
#pragma unroll
  for (int i = 0; i < 8; ++i) { a[i] = lo[i]; a[i + 8] = hi[i]; }
  return a;
}

// B fragment where k runs along the strided dimension (for V tiles in LDS):
// base = &tile[0][n-column for this lane]; element j reads tile[kpat(j)][n].
static __device__ __forceinline__ bf16x16 load_frag_kstrided(const unsigned short* base,
                                                             int stride, int lane) {
  int kb = (lane >> 4) * 8;
  bf16x16 a;
#pragma unroll
  for (int j = 0; j < 8; ++j) {
    a[j]     = us2bf(base[(size_t)(kb + j) * stride]);
    a[j + 8] = us2bf(base[(size_t)(kb + 16 + j) * stride]);
  }
  return a;
}

// reductions across 16 lanes (C-fragment: one matrix row's 16 cols live in 16 lanes)
static __device__ __forceinline__ float rmax16(float v) {
#pragma unroll
  for (int m = 1; m < 16; m <<= 1) v = fmaxf(v, __shfl_xor(v, m, 32));
  return v;
}
static __device__ __forceinline__ float rsum16(float v) {
#pragma unroll
  for (int m = 1; m < 16; m <<= 1) v += __shfl_xor(v, m, 32);
  return v;
}

// ---------------------------------------------------------------- conversions
__global__ void k_f32_to_bf16(const float* __restrict__ in,
                              unsigned short* __restrict__ out, int n) {
  int i = blockIdx.x * blockDim.x + threadIdx.x;
  if (i < n) out[i] = f2bf(in[i]);
}

// W [K=1024, Nout=1024] fp32 -> Wt [Nout, K] bf16 (so B-fragments load contiguously)
__global__ void k_transpose_bf16(const float* __restrict__ W,
                                 unsigned short* __restrict__ Wt) {
  int t = blockIdx.x * blockDim.x + threadIdx.x;  // < 1024*1024
  int n = t >> 10, k = t & 1023;
  Wt[t] = f2bf(W[(k << 10) + n]);
}

// ---------------------------------------------------------------- WMMA GEMM
// C[4096,1024] = A_bf16[4096,1024] @ Wt_bf16^T + bias
// mode 0: fp32 row-major out.  mode 1: bf16 out in [B,H,N,DH] head layout.
__global__ void k_gemm_bf16(const unsigned short* __restrict__ A,
                            const unsigned short* __restrict__ Bt,
                            const float* __restrict__ bias,
                            float* __restrict__ outF,
                            unsigned short* __restrict__ outH, int mode) {
  int lane = threadIdx.x & 31;
  int wid  = threadIdx.x >> 5;
  int tile = blockIdx.x * 8 + wid;         // 16384 tiles total
  int mt = tile >> 6;                      // 256 row tiles
  int nt = tile & 63;                      // 64 col tiles

  const unsigned short* abase = A  + (size_t)(mt * 16) * DD;
  const unsigned short* bbase = Bt + (size_t)(nt * 16) * DD;

  f32x8 acc = {};
  for (int k0 = 0; k0 < DD; k0 += 32) {
    bf16x16 af = load_frag_rowk(abase + k0, DD, lane);
    bf16x16 bf = load_frag_rowk(bbase + k0, DD, lane);
    acc = wmma_bf16(af, bf, acc);
  }

  int n  = nt * 16 + (lane & 15);
  int rb = (lane >> 4) * 8;
  float bn = bias[n];

  if (mode == 0) {
#pragma unroll
    for (int i = 0; i < 8; ++i) {
      int m = mt * 16 + rb + i;
      outF[(size_t)m * DD + n] = acc[i] + bn;
    }
  } else {
    int h = n >> 6, dh = n & 63;
#pragma unroll
    for (int i = 0; i < 8; ++i) {
      int m = mt * 16 + rb + i;
      int b = m >> 11, ns = m & 2047;
      outH[(((size_t)(b * HH + h)) * NN + ns) * DH + dh] = f2bf(acc[i] + bn);
    }
  }
}

// ---------------------------------------------------------------- attention
// One wave per (b,h, 16-query block). Streaming softmax statistics (pass 1).
__global__ void k_attn_stats(const unsigned short* __restrict__ Qb,
                             const unsigned short* __restrict__ Kb,
                             float* __restrict__ rmaxP,
                             float* __restrict__ rsumP) {
  int lane = threadIdx.x;
  int qb = blockIdx.x & 127;               // N/16
  int bh = blockIdx.x >> 7;                // B*H
  size_t bhN = (size_t)bh * NN;

  const unsigned short* Qrow = Qb + (bhN + qb * 16) * DH;
  bf16x16 qf0 = load_frag_rowk(Qrow, DH, lane);
  bf16x16 qf1 = load_frag_rowk(Qrow + 32, DH, lane);

  float m[8], l[8];
#pragma unroll
  for (int i = 0; i < 8; ++i) { m[i] = -1e30f; l[i] = 0.f; }

  int rb = (lane >> 4) * 8;
  int col = lane & 15;
  int kbmax = (qb * 16 + 15) >> 5;

  for (int kb = 0; kb <= kbmax; ++kb) {
#pragma unroll
    for (int t = 0; t < 2; ++t) {
      const unsigned short* Krow = Kb + (bhN + kb * 32 + t * 16) * DH;
      bf16x16 kf0 = load_frag_rowk(Krow, DH, lane);
      bf16x16 kf1 = load_frag_rowk(Krow + 32, DH, lane);
      f32x8 s = {};
      s = wmma_bf16(qf0, kf0, s);
      s = wmma_bf16(qf1, kf1, s);
      int key = kb * 32 + t * 16 + col;
#pragma unroll
      for (int i = 0; i < 8; ++i) {
        int q = qb * 16 + rb + i;
        float val = (key <= q) ? s[i] * SCALE : -1e30f;
        float tm = rmax16(val);
        float nm = fmaxf(m[i], tm);
        float p = __expf(val - nm);
        float ts = rsum16(p);
        l[i] = l[i] * __expf(m[i] - nm) + ts;
        m[i] = nm;
      }
    }
  }
  if (col == 0) {
#pragma unroll
    for (int i = 0; i < 8; ++i) {
      int q = qb * 16 + rb + i;
      rmaxP[bhN + q] = m[i];
      rsumP[bhN + q] = l[i];
    }
  }
}

// Pass 2: recompute scores, emit A (full row incl. zeros), accumulate y = P @ V.
__global__ void k_attn_out(const unsigned short* __restrict__ Qb,
                           const unsigned short* __restrict__ Kb,
                           const unsigned short* __restrict__ Vb,
                           const float* __restrict__ rmaxP,
                           const float* __restrict__ rsumP,
                           float* __restrict__ Aout,
                           unsigned short* __restrict__ Yattn) {
  __shared__ __align__(16) unsigned short Vtile[32][72];  // 32 keys x 64 dh (+pad)
  __shared__ __align__(16) unsigned short Ptile[16][40];  // 16 q x 32 keys (+pad)

  int lane = threadIdx.x;
  int qb = blockIdx.x & 127;
  int bh = blockIdx.x >> 7;
  size_t bhN = (size_t)bh * NN;
  size_t Abase = (size_t)bh * NN * NN;

  const unsigned short* Qrow = Qb + (bhN + qb * 16) * DH;
  bf16x16 qf0 = load_frag_rowk(Qrow, DH, lane);
  bf16x16 qf1 = load_frag_rowk(Qrow + 32, DH, lane);

  int rb = (lane >> 4) * 8;
  int col = lane & 15;

  float mi[8], li[8];
#pragma unroll
  for (int i = 0; i < 8; ++i) {
    int q = qb * 16 + rb + i;
    mi[i] = rmaxP[bhN + q];
    li[i] = 1.0f / rsumP[bhN + q];
  }

  f32x8 y0 = {}, y1 = {}, y2 = {}, y3 = {};
  int kbmax = (qb * 16 + 15) >> 5;

  for (int kb = 0; kb < NN / 32; ++kb) {
    if (kb > kbmax) {  // fully masked: A = 0 (d_out is poisoned, must be written)
#pragma unroll
      for (int t = 0; t < 2; ++t)
#pragma unroll
        for (int i = 0; i < 8; ++i) {
          int q = qb * 16 + rb + i;
          Aout[Abase + (size_t)q * NN + kb * 32 + t * 16 + col] = 0.f;
        }
      continue;
    }

    // stage V key-block tile into LDS (lane -> one key row, 128B vectorized)
    {
      const uint4* src = (const uint4*)(Vb + (bhN + kb * 32 + lane) * DH);
      uint4* dst = (uint4*)(&Vtile[lane][0]);
#pragma unroll
      for (int i = 0; i < 4; ++i) dst[i] = src[i];
#pragma unroll
      for (int i = 4; i < 8; ++i) dst[i] = src[i];
    }

#pragma unroll
    for (int t = 0; t < 2; ++t) {
      const unsigned short* Krow = Kb + (bhN + kb * 32 + t * 16) * DH;
      bf16x16 kf0 = load_frag_rowk(Krow, DH, lane);
      bf16x16 kf1 = load_frag_rowk(Krow + 32, DH, lane);
      f32x8 s = {};
      s = wmma_bf16(qf0, kf0, s);
      s = wmma_bf16(qf1, kf1, s);
      int key = kb * 32 + t * 16 + col;
#pragma unroll
      for (int i = 0; i < 8; ++i) {
        int q = qb * 16 + rb + i;
        float val = (key <= q) ? s[i] * SCALE : -1e30f;
        float p = __expf(val - mi[i]) * li[i];
        Aout[Abase + (size_t)q * NN + key] = p;
        Ptile[rb + i][t * 16 + col] = f2bf(p);
      }
    }
    __syncthreads();

    bf16x16 pf = load_frag_rowk(&Ptile[0][0], 40, lane);
    bf16x16 v0 = load_frag_kstrided(&Vtile[0][0 * 16 + col], 72, lane);
    bf16x16 v1 = load_frag_kstrided(&Vtile[0][1 * 16 + col], 72, lane);
    bf16x16 v2 = load_frag_kstrided(&Vtile[0][2 * 16 + col], 72, lane);
    bf16x16 v3 = load_frag_kstrided(&Vtile[0][3 * 16 + col], 72, lane);
    y0 = wmma_bf16(pf, v0, y0);
    y1 = wmma_bf16(pf, v1, y1);
    y2 = wmma_bf16(pf, v2, y2);
    y3 = wmma_bf16(pf, v3, y3);
    __syncthreads();
  }

  // write y tile to [B,N,D] bf16 buffer for the projection GEMM
  int b = bh >> 4, h = bh & 15;
#pragma unroll
  for (int i = 0; i < 8; ++i) {
    int q = qb * 16 + rb + i;
    size_t base = ((size_t)b * NN + q) * DD + h * DH;
    Yattn[base + 0 * 16 + col] = f2bf(y0[i]);
    Yattn[base + 1 * 16 + col] = f2bf(y1[i]);
    Yattn[base + 2 * 16 + col] = f2bf(y2[i]);
    Yattn[base + 3 * 16 + col] = f2bf(y3[i]);
  }
}

// ---------------------------------------------------------------- launch
extern "C" void kernel_launch(void* const* d_in, const int* in_sizes, int n_in,
                              void* d_out, int out_size, void* d_ws, size_t ws_size,
                              hipStream_t stream) {
  const float* x  = (const float*)d_in[0];
  // d_in[1] is the causal mask (int32) — causality is hardcoded below.
  const float* Wq = (const float*)d_in[2]; const float* bq = (const float*)d_in[3];
  const float* Wk = (const float*)d_in[4]; const float* bk = (const float*)d_in[5];
  const float* Wv = (const float*)d_in[6]; const float* bv = (const float*)d_in[7];
  const float* Wp = (const float*)d_in[8]; const float* bp = (const float*)d_in[9];

  float* out  = (float*)d_out;
  float* Aout = out + (size_t)BB * NN * DD;  // tuple order: (y, A)

  char* ws = (char*)d_ws;
  const size_t M = (size_t)BB * NN;          // 4096
  unsigned short* xb  = (unsigned short*)ws;                 ws += M * DD * 2;       // 8 MB
  unsigned short* WqT = (unsigned short*)ws;                 ws += (size_t)DD * DD * 2;
  unsigned short* WkT = (unsigned short*)ws;                 ws += (size_t)DD * DD * 2;
  unsigned short* WvT = (unsigned short*)ws;                 ws += (size_t)DD * DD * 2;
  unsigned short* WpT = (unsigned short*)ws;                 ws += (size_t)DD * DD * 2;
  unsigned short* Qb  = (unsigned short*)ws;                 ws += M * DD * 2;
  unsigned short* Kb  = (unsigned short*)ws;                 ws += M * DD * 2;
  unsigned short* Vb  = (unsigned short*)ws;                 ws += M * DD * 2;
  float* rmaxP        = (float*)ws;                          ws += (size_t)BB * HH * NN * 4;
  float* rsumP        = (float*)ws;                          ws += (size_t)BB * HH * NN * 4;
  unsigned short* Yat = (unsigned short*)ws;                 ws += M * DD * 2;

  // 1) fp32 -> bf16 conversions / weight transposes
  k_f32_to_bf16<<<(int)(M * DD / 256), 256, 0, stream>>>(x, xb, (int)(M * DD));
  k_transpose_bf16<<<DD * DD / 256, 256, 0, stream>>>(Wq, WqT);
  k_transpose_bf16<<<DD * DD / 256, 256, 0, stream>>>(Wk, WkT);
  k_transpose_bf16<<<DD * DD / 256, 256, 0, stream>>>(Wv, WvT);
  k_transpose_bf16<<<DD * DD / 256, 256, 0, stream>>>(Wp, WpT);

  // 2) Q/K/V projections (WMMA), bf16 head-layout outputs
  const int gemm_blocks = (int)(M / 16) * (DD / 16) / 8;  // 2048
  k_gemm_bf16<<<gemm_blocks, 256, 0, stream>>>(xb, WqT, bq, nullptr, Qb, 1);
  k_gemm_bf16<<<gemm_blocks, 256, 0, stream>>>(xb, WkT, bk, nullptr, Kb, 1);
  k_gemm_bf16<<<gemm_blocks, 256, 0, stream>>>(xb, WvT, bv, nullptr, Vb, 1);

  // 3) attention: softmax statistics, then A + y
  const int attn_blocks = BB * HH * (NN / 16);  // 4096
  k_attn_stats<<<attn_blocks, 32, 0, stream>>>(Qb, Kb, rmaxP, rsumP);
  k_attn_out<<<attn_blocks, 32, 0, stream>>>(Qb, Kb, Vb, rmaxP, rsumP, Aout, Yat);

  // 4) output projection (WMMA), fp32 output into d_out
  k_gemm_bf16<<<gemm_blocks, 256, 0, stream>>>(Yat, WpT, bp, out, nullptr, 0);
}
